// GeoAwarePooling_35296041238610
// MI455X (gfx1250) — matrix operand order repaired and checked by reading, not compiled, not run on hardware
//
#include <hip/hip_runtime.h>
#include <math.h>

typedef __attribute__((ext_vector_type(16))) _Float16 v16h;
typedef __attribute__((ext_vector_type(8)))  _Float16 v8h;
typedef __attribute__((ext_vector_type(8)))  float    v8f;

#define WAVES  4
#define TPB    (WAVES * 32)
#define NFRAG  30
#define GRID_A 512
#define GRID_B 512
#define GRID_C 512

// ---------------------------------------------------------------------------
// helpers
// ---------------------------------------------------------------------------
__device__ __forceinline__ void lds_wave_fence() {
  // CDNA5 split wait counters: wait for all outstanding LDS ops of this wave
  asm volatile("s_wait_dscnt 0" ::: "memory");
}

__device__ __forceinline__ v8f vzero8() {
  v8f z;
#pragma unroll
  for (int i = 0; i < 8; ++i) z[i] = 0.f;
  return z;
}

// Build one 16x32 f16 A-fragment from a 16x128-half LDS tile (row-major).
// This code's k order (same as B): lane l holds M=l&15; K(e) = (l<16?0:16) + e
// => elements 0..7 and 8..15 are each 8 contiguous halves -> two b128 loads.
__device__ __forceinline__ v16h load_a_frag(const _Float16* rows, int kt, int lane) {
  int m  = lane & 15;
  int kb = (lane < 16) ? 0 : 16;
  const _Float16* base = rows + m * 128 + kt * 32 + kb;
  v8h lo = *(const v8h*)(base);
  v8h hi = *(const v8h*)(base + 8);
  return __builtin_shufflevector(lo, hi, 0,1,2,3,4,5,6,7,8,9,10,11,12,13,14,15);
}

// LayerNorm over 64 channels laid out in 4 C/D fragments (v8f each).
// C/D layout: lane l holds (M = r + (l<16?0:8), N = l&15) for vgpr r.
// Sum over channels == sum over 4 tiles + reduction across each 16-lane half.
__device__ __forceinline__ void layernorm64(const v8f acc[4], const float* bias,
                                            const float* g, const float* be,
                                            v8f out[4], bool relu) {
#pragma unroll
  for (int r = 0; r < 8; ++r) {
    float x[4];
    float s = 0.f, q = 0.f;
#pragma unroll
    for (int nt = 0; nt < 4; ++nt) {
      x[nt] = acc[nt][r] + bias[nt];
      s += x[nt];
      q += x[nt] * x[nt];
    }
#pragma unroll
    for (int msk = 1; msk < 16; msk <<= 1) {   // stays within 16-lane halves
      s += __shfl_xor(s, msk, 32);
      q += __shfl_xor(q, msk, 32);
    }
    float mean = s * (1.f / 64.f);
    float var  = q * (1.f / 64.f) - mean * mean;
    float rstd = rsqrtf(var + 1e-5f);
#pragma unroll
    for (int nt = 0; nt < 4; ++nt) {
      float y = (x[nt] - mean) * rstd * g[nt] + be[nt];
      if (relu) y = fmaxf(y, 0.f);
      out[nt][r] = y;
    }
  }
}

// Pre-swizzle weight matrices into WMMA B-fragment layout in LDS.
// B layout (16-bit, KxN=32x16): lane l: N=l&15, K(e) = (l<16?0:16)+e (+kt*32).
// frag 0..3  : layer1 (w1 3x64, zero-padded to K=32), nt = f
// frag 4..11 : layer2 (w2 64x64), kt=(f-4)>>2, nt=(f-4)&3
// frag 12..27: layer3 (w3 128x64), kt=(f-12)>>2, nt=(f-12)&3
// frag 28..29: layer4 (w4 64x1 in column 0), kt=f-28
__device__ __forceinline__ void fill_swz(_Float16 (*swz)[32][16], int nfrags,
                                         const float* w1, const float* w2,
                                         const float* w3, const float* w4) {
  for (int idx = threadIdx.x; idx < nfrags * 32; idx += blockDim.x) {
    int f = idx >> 5, l = idx & 31;
    int nlo  = l & 15;
    int koff = (l < 16) ? 0 : 16;
#pragma unroll
    for (int e = 0; e < 16; ++e) {
      int k = koff + e;
      float v = 0.f;
      if (f < 4) {
        if (k < 3) v = w1[k * 64 + (f * 16 + nlo)];
      } else if (f < 12) {
        int q = f - 4;
        v = w2[((q >> 2) * 32 + k) * 64 + ((q & 3) * 16 + nlo)];
      } else if (f < 28) {
        int q = f - 12;
        v = w3[((q >> 2) * 32 + k) * 64 + ((q & 3) * 16 + nlo)];
      } else {
        if (nlo == 0) v = w4[(f - 28) * 32 + k];
      }
      swz[f][l][e] = (_Float16)v;
    }
  }
}

// Layers 1+2 for one 16-point tile -> acc2 = xyz_feat (LN2 output, D-layout).
// B fragments for layers 1 and 2 are loop-invariant and held in VGPRs.
__device__ __forceinline__ void compute_xyzfeat(
    int p0, int N, const float* __restrict__ xyz,
    float cx, float cy, float cz, float inv,
    const v16h* bw1, const v16h* bw2, _Float16* catw,
    const float* b1r, const float* g1r, const float* be1r,
    const float* b2r, const float* g2r, const float* be2r,
    int lane, v8f acc2[4]) {
  // layer1 A fragment: only lanes<16 carry K=0..2 (x,y,z), everything else 0
  v16h a1;
#pragma unroll
  for (int e = 0; e < 16; ++e) a1[e] = (_Float16)0.f;
  if (lane < 16) {
    int p = p0 + lane;
    if (p >= N) p = N - 1;
    a1[0] = (_Float16)((xyz[p * 3 + 0] - cx) * inv);
    a1[1] = (_Float16)((xyz[p * 3 + 1] - cy) * inv);
    a1[2] = (_Float16)((xyz[p * 3 + 2] - cz) * inv);
  }
  v8f acc1[4];
#pragma unroll
  for (int nt = 0; nt < 4; ++nt)
    acc1[nt] = __builtin_amdgcn_wmma_f32_16x16x32_f16(false, a1, false, bw1[nt],
                                                      (short)0, vzero8(), false, false);
  v8f h[4];
  layernorm64(acc1, b1r, g1r, be1r, h, true);

  // stage h (f16) in LDS point-major for the layer2 A fragments
  int nlo = lane & 15, mofs = (lane < 16) ? 0 : 8;
#pragma unroll
  for (int nt = 0; nt < 4; ++nt)
#pragma unroll
    for (int r = 0; r < 8; ++r)
      catw[(r + mofs) * 128 + nt * 16 + nlo] = (_Float16)h[nt][r];
  lds_wave_fence();

#pragma unroll
  for (int nt = 0; nt < 4; ++nt) acc2[nt] = vzero8();
#pragma unroll
  for (int kt = 0; kt < 2; ++kt) {
    v16h a = load_a_frag(catw, kt, lane);
#pragma unroll
    for (int nt = 0; nt < 4; ++nt)
      acc2[nt] = __builtin_amdgcn_wmma_f32_16x16x32_f16(false, a, false, bw2[kt * 4 + nt],
                                                        (short)0, acc2[nt], false, false);
  }
  v8f t[4];
  layernorm64(acc2, b2r, g2r, be2r, t, false);
#pragma unroll
  for (int nt = 0; nt < 4; ++nt) acc2[nt] = t[nt];
}

// ---------------------------------------------------------------------------
// Pass A: xyz min/max/sum
// ---------------------------------------------------------------------------
__global__ __launch_bounds__(256) void ga_kA(const float* __restrict__ xyz,
                                             float* __restrict__ partA, int N) {
  float mn[3] = {3.4e38f, 3.4e38f, 3.4e38f};
  float mx[3] = {-3.4e38f, -3.4e38f, -3.4e38f};
  float sm[3] = {0.f, 0.f, 0.f};
  for (int i = blockIdx.x * 256 + threadIdx.x; i < N; i += gridDim.x * 256) {
#pragma unroll
    for (int c = 0; c < 3; ++c) {
      float v = xyz[i * 3 + c];
      mn[c] = fminf(mn[c], v);
      mx[c] = fmaxf(mx[c], v);
      sm[c] += v;
    }
  }
  __shared__ float sr[256][9];
#pragma unroll
  for (int c = 0; c < 3; ++c) {
    sr[threadIdx.x][c] = mn[c];
    sr[threadIdx.x][3 + c] = mx[c];
    sr[threadIdx.x][6 + c] = sm[c];
  }
  __syncthreads();
  for (int s = 128; s > 0; s >>= 1) {
    if (threadIdx.x < (unsigned)s) {
#pragma unroll
      for (int c = 0; c < 3; ++c) {
        sr[threadIdx.x][c]     = fminf(sr[threadIdx.x][c],     sr[threadIdx.x + s][c]);
        sr[threadIdx.x][3 + c] = fmaxf(sr[threadIdx.x][3 + c], sr[threadIdx.x + s][3 + c]);
        sr[threadIdx.x][6 + c] += sr[threadIdx.x + s][6 + c];
      }
    }
    __syncthreads();
  }
  if (threadIdx.x == 0)
#pragma unroll
    for (int k = 0; k < 9; ++k) partA[blockIdx.x * 9 + k] = sr[0][k];
}

__global__ __launch_bounds__(256) void ga_kA2(const float* __restrict__ partA,
                                              float* __restrict__ hdr, int nblk, int N) {
  float mn[3] = {3.4e38f, 3.4e38f, 3.4e38f};
  float mx[3] = {-3.4e38f, -3.4e38f, -3.4e38f};
  float sm[3] = {0.f, 0.f, 0.f};
  for (int b = threadIdx.x; b < nblk; b += 256) {
#pragma unroll
    for (int c = 0; c < 3; ++c) {
      mn[c] = fminf(mn[c], partA[b * 9 + c]);
      mx[c] = fmaxf(mx[c], partA[b * 9 + 3 + c]);
      sm[c] += partA[b * 9 + 6 + c];
    }
  }
  __shared__ float sr[256][9];
#pragma unroll
  for (int c = 0; c < 3; ++c) {
    sr[threadIdx.x][c] = mn[c];
    sr[threadIdx.x][3 + c] = mx[c];
    sr[threadIdx.x][6 + c] = sm[c];
  }
  __syncthreads();
  for (int s = 128; s > 0; s >>= 1) {
    if (threadIdx.x < (unsigned)s) {
#pragma unroll
      for (int c = 0; c < 3; ++c) {
        sr[threadIdx.x][c]     = fminf(sr[threadIdx.x][c],     sr[threadIdx.x + s][c]);
        sr[threadIdx.x][3 + c] = fmaxf(sr[threadIdx.x][3 + c], sr[threadIdx.x + s][3 + c]);
        sr[threadIdx.x][6 + c] += sr[threadIdx.x + s][6 + c];
      }
    }
    __syncthreads();
  }
  if (threadIdx.x == 0) {
    float dia = fmaxf(fmaxf(sr[0][3] - sr[0][0], sr[0][4] - sr[0][1]), sr[0][5] - sr[0][2]);
    float invN = 1.f / (float)N;
    hdr[0] = sr[0][6] * invN;
    hdr[1] = sr[0][7] * invN;
    hdr[2] = sr[0][8] * invN;
    hdr[3] = 1.f / (dia + 1e-4f);
  }
}

// ---------------------------------------------------------------------------
// Pass B: xyz_feat = LN2(LN1(xyz@w1+b1).relu @ w2 + b2); per-block channel max
// ---------------------------------------------------------------------------
__global__ __launch_bounds__(TPB) void ga_kB(
    const float* __restrict__ xyz,
    const float* w1, const float* b1, const float* g1, const float* be1,
    const float* w2, const float* b2, const float* g2, const float* be2,
    const float* __restrict__ hdr, float* __restrict__ partB, int N, int ntiles) {
  __shared__ __align__(32) _Float16 swz[12][32][16];
  __shared__ __align__(32) _Float16 cat[WAVES][16][128];
  __shared__ float red[WAVES][64];
  fill_swz(swz, 12, w1, w2, w2, w2);
  __syncthreads();

  float cx = hdr[0], cy = hdr[1], cz = hdr[2], inv = hdr[3];
  int lane = threadIdx.x & 31, wave = threadIdx.x >> 5;
  int nlo = lane & 15;

  // hoist loop-invariant B fragments into VGPRs (12 frags = 96 VGPRs)
  v16h bw1[4], bw2[8];
#pragma unroll
  for (int i = 0; i < 4; ++i) bw1[i] = *(const v16h*)(&swz[i][lane][0]);
#pragma unroll
  for (int i = 0; i < 8; ++i) bw2[i] = *(const v16h*)(&swz[4 + i][lane][0]);

  float b1r[4], g1r[4], be1r[4], b2r[4], g2r[4], be2r[4];
#pragma unroll
  for (int nt = 0; nt < 4; ++nt) {
    int ch = nt * 16 + nlo;
    b1r[nt] = b1[ch]; g1r[nt] = g1[ch]; be1r[nt] = be1[ch];
    b2r[nt] = b2[ch]; g2r[nt] = g2[ch]; be2r[nt] = be2[ch];
  }
  float cmax[4] = {-3.4e38f, -3.4e38f, -3.4e38f, -3.4e38f};
  int gw = blockIdx.x * WAVES + wave, nw = gridDim.x * WAVES;
  for (int t = gw; t < ntiles; t += nw) {
    v8f acc2[4];
    compute_xyzfeat(t * 16, N, xyz, cx, cy, cz, inv, bw1, bw2, &cat[wave][0][0],
                    b1r, g1r, be1r, b2r, g2r, be2r, lane, acc2);
#pragma unroll
    for (int nt = 0; nt < 4; ++nt)
#pragma unroll
      for (int r = 0; r < 8; ++r) cmax[nt] = fmaxf(cmax[nt], acc2[nt][r]);
  }
#pragma unroll
  for (int nt = 0; nt < 4; ++nt)
    cmax[nt] = fmaxf(cmax[nt], __shfl_xor(cmax[nt], 16, 32));
  if (lane < 16)
#pragma unroll
    for (int nt = 0; nt < 4; ++nt) red[wave][nt * 16 + nlo] = cmax[nt];
  __syncthreads();
  if (threadIdx.x < 64) {
    float m = red[0][threadIdx.x];
#pragma unroll
    for (int w = 1; w < WAVES; ++w) m = fmaxf(m, red[w][threadIdx.x]);
    partB[blockIdx.x * 64 + threadIdx.x] = m;
  }
}

__global__ __launch_bounds__(64) void ga_kB2(const float* __restrict__ partB,
                                             float* __restrict__ glb, int nblk) {
  float m = -3.4e38f;
  for (int b = 0; b < nblk; ++b) m = fmaxf(m, partB[b * 64 + threadIdx.x]);
  glb[threadIdx.x] = m;
}

// ---------------------------------------------------------------------------
// Pass C: recompute xyz_feat, layer3+4, sigmoid weight, feat accumulation
// ---------------------------------------------------------------------------
__global__ __launch_bounds__(TPB) void ga_kC(
    const float* __restrict__ feat, const float* __restrict__ xyz,
    const float* w1, const float* b1, const float* g1, const float* be1,
    const float* w2, const float* b2, const float* g2, const float* be2,
    const float* w3, const float* b3, const float* g3, const float* be3,
    const float* w4,
    const float* __restrict__ hdr, const float* __restrict__ glb,
    float* __restrict__ partC, int N, int ntiles) {
  __shared__ __align__(32) _Float16 swz[NFRAG][32][16];
  __shared__ __align__(32) _Float16 cat[WAVES][16][128];
  __shared__ float red[WAVES][256];
  __shared__ float wv[WAVES][16];
  fill_swz(swz, NFRAG, w1, w2, w3, w4);
  __syncthreads();

  float cx = hdr[0], cy = hdr[1], cz = hdr[2], inv = hdr[3];
  int lane = threadIdx.x & 31, wave = threadIdx.x >> 5;
  int nlo = lane & 15, mofs = (lane < 16) ? 0 : 8;

  // hoist loop-invariant B fragments for layers 1, 2 and 4 into VGPRs
  v16h bw1[4], bw2[8], bw4[2];
#pragma unroll
  for (int i = 0; i < 4; ++i) bw1[i] = *(const v16h*)(&swz[i][lane][0]);
#pragma unroll
  for (int i = 0; i < 8; ++i) bw2[i] = *(const v16h*)(&swz[4 + i][lane][0]);
#pragma unroll
  for (int i = 0; i < 2; ++i) bw4[i] = *(const v16h*)(&swz[28 + i][lane][0]);

  float b1r[4], g1r[4], be1r[4], b2r[4], g2r[4], be2r[4], b3r[4], g3r[4], be3r[4];
#pragma unroll
  for (int nt = 0; nt < 4; ++nt) {
    int ch = nt * 16 + nlo;
    b1r[nt] = b1[ch]; g1r[nt] = g1[ch]; be1r[nt] = be1[ch];
    b2r[nt] = b2[ch]; g2r[nt] = g2[ch]; be2r[nt] = be2[ch];
    b3r[nt] = b3[ch]; g3r[nt] = g3[ch]; be3r[nt] = be3[ch];
  }
  // prefill the broadcast (glb) half of this wave's concat tile
  for (int i = lane; i < 16 * 64; i += 32)
    cat[wave][i >> 6][64 + (i & 63)] = (_Float16)glb[i & 63];

  float fsum[8];
#pragma unroll
  for (int k = 0; k < 8; ++k) fsum[k] = 0.f;

  int gw = blockIdx.x * WAVES + wave, nw = gridDim.x * WAVES;
  for (int t = gw; t < ntiles; t += nw) {
    int p0 = t * 16;
    v8f acc2[4];
    compute_xyzfeat(p0, N, xyz, cx, cy, cz, inv, bw1, bw2, &cat[wave][0][0],
                    b1r, g1r, be1r, b2r, g2r, be2r, lane, acc2);
    // write xyz_feat into concat cols 0..63 (cols 64..127 already hold glb)
#pragma unroll
    for (int nt = 0; nt < 4; ++nt)
#pragma unroll
      for (int r = 0; r < 8; ++r)
        cat[wave][r + mofs][nt * 16 + nlo] = (_Float16)acc2[nt][r];
    lds_wave_fence();

    // layer3: K = 128 (B fragments from LDS; too many to hold in VGPRs)
    v8f acc3[4];
#pragma unroll
    for (int nt = 0; nt < 4; ++nt) acc3[nt] = vzero8();
#pragma unroll
    for (int kt = 0; kt < 4; ++kt) {
      v16h a = load_a_frag(&cat[wave][0][0], kt, lane);
#pragma unroll
      for (int nt = 0; nt < 4; ++nt) {
        v16h b = *(const v16h*)(&swz[12 + kt * 4 + nt][lane][0]);
        acc3[nt] = __builtin_amdgcn_wmma_f32_16x16x32_f16(false, a, false, b,
                                                          (short)0, acc3[nt], false, false);
      }
    }
    v8f h2[4];
    layernorm64(acc3, b3r, g3r, be3r, h2, true);
    // stage h2 for the layer4 matmul
#pragma unroll
    for (int nt = 0; nt < 4; ++nt)
#pragma unroll
      for (int r = 0; r < 8; ++r)
        cat[wave][r + mofs][nt * 16 + nlo] = (_Float16)h2[nt][r];
    lds_wave_fence();

    // layer4: h2 @ w4 (w4 lives in column 0 of the B fragments)
    v8f acc4 = vzero8();
#pragma unroll
    for (int kt = 0; kt < 2; ++kt) {
      v16h a = load_a_frag(&cat[wave][0][0], kt, lane);
      acc4 = __builtin_amdgcn_wmma_f32_16x16x32_f16(false, a, false, bw4[kt],
                                                    (short)0, acc4, false, false);
    }
    if (nlo == 0) {
#pragma unroll
      for (int r = 0; r < 8; ++r)
        wv[wave][r + mofs] = 2.f / (1.f + expf(-acc4[r]));
    }
    lds_wave_fence();

    // stream feat: each lane owns channels [lane*4..+3] and [128+lane*4..+3]
    for (int m = 0; m < 16; ++m) {
      int p = p0 + m;
      if (p >= N) break;
      float wgt = wv[wave][m];
      const float4* fp = reinterpret_cast<const float4*>(feat + (size_t)p * 256);
      float4 fa = fp[lane];
      float4 fb = fp[lane + 32];
      fsum[0] += fa.x * wgt; fsum[1] += fa.y * wgt;
      fsum[2] += fa.z * wgt; fsum[3] += fa.w * wgt;
      fsum[4] += fb.x * wgt; fsum[5] += fb.y * wgt;
      fsum[6] += fb.z * wgt; fsum[7] += fb.w * wgt;
    }
  }
#pragma unroll
  for (int k = 0; k < 4; ++k) {
    red[wave][lane * 4 + k]       = fsum[k];
    red[wave][128 + lane * 4 + k] = fsum[4 + k];
  }
  __syncthreads();
  for (int ch = threadIdx.x; ch < 256; ch += TPB) {
    float s = 0.f;
#pragma unroll
    for (int w = 0; w < WAVES; ++w) s += red[w][ch];
    partC[blockIdx.x * 256 + ch] = s;
  }
}

__global__ __launch_bounds__(256) void ga_kC2(const float* __restrict__ partC,
                                              float* __restrict__ out, int nblk, int N) {
  float s = 0.f;
  for (int b = 0; b < nblk; ++b) s += partC[b * 256 + threadIdx.x];
  out[threadIdx.x] = s / (float)N;
}

// ---------------------------------------------------------------------------
extern "C" void kernel_launch(void* const* d_in, const int* in_sizes, int n_in,
                              void* d_out, int out_size, void* d_ws, size_t ws_size,
                              hipStream_t stream) {
  const float* feat = (const float*)d_in[0];
  const float* xyz  = (const float*)d_in[1];
  const float* w1   = (const float*)d_in[2];
  const float* b1   = (const float*)d_in[3];
  const float* g1   = (const float*)d_in[4];
  const float* be1  = (const float*)d_in[5];
  const float* w2   = (const float*)d_in[6];
  const float* b2   = (const float*)d_in[7];
  const float* g2   = (const float*)d_in[8];
  const float* be2  = (const float*)d_in[9];
  const float* w3   = (const float*)d_in[10];
  const float* b3   = (const float*)d_in[11];
  const float* g3   = (const float*)d_in[12];
  const float* be3  = (const float*)d_in[13];
  const float* w4   = (const float*)d_in[14];

  int N = in_sizes[1] / 3;
  int ntiles = (N + 15) / 16;

  float* ws  = (float*)d_ws;
  float* hdr = ws;                      // 4 floats: center xyz, 1/(dia+eps)
  float* glb = ws + 4;                  // 64 floats
  float* pA  = ws + 68;                 // GRID_A * 9
  float* pB  = pA + GRID_A * 9;         // GRID_B * 64
  float* pC  = pB + GRID_B * 64;        // GRID_C * 256

  ga_kA <<<GRID_A, 256, 0, stream>>>(xyz, pA, N);
  ga_kA2<<<1, 256, 0, stream>>>(pA, hdr, GRID_A, N);
  ga_kB <<<GRID_B, TPB, 0, stream>>>(xyz, w1, b1, g1, be1, w2, b2, g2, be2,
                                     hdr, pB, N, ntiles);
  ga_kB2<<<1, 64, 0, stream>>>(pB, glb, GRID_B);
  ga_kC <<<GRID_C, TPB, 0, stream>>>(feat, xyz, w1, b1, g1, be1, w2, b2, g2, be2,
                                     w3, b3, g3, be3, w4, hdr, glb, pC, N, ntiles);
  ga_kC2<<<1, 256, 0, stream>>>(pC, (float*)d_out, GRID_C, N);
}
